// JiTAttention_58420145160236
// MI455X (gfx1250) — compile-verified, matrix operand order checked by reading, not run on hardware
//
#include <hip/hip_runtime.h>
#include <stdint.h>

// ---------------------------------------------------------------------------
// Types for CDNA5 WMMA (wave32): bf16 16x16x32, f32 accum
// ---------------------------------------------------------------------------
typedef __attribute__((ext_vector_type(16))) __bf16 v16bf;
typedef __attribute__((ext_vector_type(8)))  float  v8f;

struct __align__(16) U4 { unsigned int x, y, z, w; };

union FragU {
  U4    u[2];
  v16bf v;
};

__device__ __forceinline__ unsigned short f2bf(float f) {
  unsigned int u = __float_as_uint(f);
  u += 0x7FFFu + ((u >> 16) & 1u);          // round-to-nearest-even
  return (unsigned short)(u >> 16);
}
__device__ __forceinline__ float bf2f(unsigned short h) {
  return __uint_as_float((unsigned int)h << 16);
}

__device__ __forceinline__ v8f bf16_wmma(const FragU& a, const FragU& b, v8f c) {
  return __builtin_amdgcn_wmma_f32_16x16x32_bf16(false, a.v, false, b.v,
                                                 (short)0, c, false, false);
}

// ---------------------------------------------------------------------------
// CDNA5 async global->LDS DMA (ASYNCcnt-tracked). The low 32 bits of a
// generic pointer into __shared__ are the LDS byte offset.
// ---------------------------------------------------------------------------
__device__ __forceinline__ void async_b128(void* lds_ptr, const void* gptr) {
  unsigned int off = (unsigned int)(size_t)lds_ptr;
  asm volatile("global_load_async_to_lds_b128 %0, %1, off"
               :: "v"(off), "v"(gptr) : "memory");
}
#define WAIT_ASYNC(n) asm volatile("s_wait_asynccnt " #n ::: "memory")

// ---------------------------------------------------------------------------
// 1) f32 -> bf16 conversion
// ---------------------------------------------------------------------------
__global__ void cvt_f32_bf16(const float* __restrict__ in,
                             unsigned short* __restrict__ out, int n) {
  int i = blockIdx.x * blockDim.x + threadIdx.x;
  if (i < n) out[i] = f2bf(in[i]);
}

// ---------------------------------------------------------------------------
// 2/5) bf16 WMMA GEMM:  out[m,n] = sum_k A[m,k]*W[n,k] + bias[n]
//  A: [M,K] row-major bf16, W: [Nout,K] row-major bf16.
//  Block = 256 threads = 8 waves -> 64(M) x 128(N) tile; wave = 16x64.
//  Double-buffered async-to-LDS tile pipeline (3 async ops/thread/tile).
//  MODE 0: out = f32 [M,Nout]  (proj)
//  MODE 1: out = bf16 scattered; q,k -> [s,B,H,N,64], v -> transposed [B,H,64,N]
// ---------------------------------------------------------------------------
template <int MODE>
__global__ __launch_bounds__(256) void gemm_bf16_kernel(
    const unsigned short* __restrict__ A,
    const unsigned short* __restrict__ W,
    const float* __restrict__ bias,
    void* __restrict__ out,
    int K, int Nout)
{
  __shared__ unsigned short As[2][64 * 32];
  __shared__ unsigned short Ws[2][128 * 32];

  const int tid  = threadIdx.x;
  const int lane = tid & 31, wave = tid >> 5;
  const int half = lane >> 4, nl = lane & 15, kA = half * 8;
  const int waveN = wave & 1, waveM = wave >> 1;
  const int mbase = blockIdx.y * 64, nbase = blockIdx.x * 128;

  v8f zero = {};
  v8f c[4] = { zero, zero, zero, zero };

  const int lrow = tid >> 2, lch = (tid & 3) * 8;

  auto issue = [&](int k0, int buf) {
    async_b128(&As[buf][lrow * 32 + lch],        &A[(size_t)(mbase + lrow) * K + k0 + lch]);
    async_b128(&Ws[buf][lrow * 32 + lch],        &W[(size_t)(nbase + lrow) * K + k0 + lch]);
    async_b128(&Ws[buf][(lrow + 64) * 32 + lch], &W[(size_t)(nbase + lrow + 64) * K + k0 + lch]);
  };

  issue(0, 0);
  int cur = 0;
  for (int k0 = 0; k0 < K; k0 += 32) {
    __syncthreads();                       // readers of buf^1 (prev iter) done
    if (k0 + 32 < K) {
      issue(k0 + 32, cur ^ 1);             // prefetch next tile
      WAIT_ASYNC(0x3);                     // current tile (3 ops) complete
    } else {
      WAIT_ASYNC(0x0);
    }
    __syncthreads();                       // publish buf[cur]

    FragU a;
    const int ar = waveM * 16 + nl;
    a.u[0] = *(const U4*)&As[cur][ar * 32 + kA];
    a.u[1] = *(const U4*)&As[cur][ar * 32 + kA + 16];
#pragma unroll
    for (int dt = 0; dt < 4; ++dt) {
      FragU b;
      const int br = waveN * 64 + dt * 16 + nl;
      b.u[0] = *(const U4*)&Ws[cur][br * 32 + kA];
      b.u[1] = *(const U4*)&Ws[cur][br * 32 + kA + 16];
      c[dt] = bf16_wmma(a, b, c[dt]);
    }
    cur ^= 1;
  }

#pragma unroll
  for (int dt = 0; dt < 4; ++dt) {
#pragma unroll
    for (int i = 0; i < 8; ++i) {
      const int n = nbase + waveN * 64 + dt * 16 + nl;
      const int m = mbase + waveM * 16 + i + 8 * half;
      const float val = c[dt][i] + bias[n];
      if (MODE == 0) {
        ((float*)out)[(size_t)m * Nout + n] = val;
      } else {
        const int s = n >> 10, rr = n & 1023, h = rr >> 6, d = rr & 63;
        const int b = m >> 11, t = m & 2047;
        if (s == 2) {  // V stored transposed: [b,h,d,t] for async attention loads
          ((unsigned short*)out)[(size_t)2 * 8388608 +
              (((size_t)(b * 16 + h) * 64 + d) * 2048 + t)] = f2bf(val);
        } else {
          ((unsigned short*)out)[((((size_t)s * 4 + b) * 16 + h) * 2048 + t) * 64 + d] = f2bf(val);
        }
      }
    }
  }
}

// ---------------------------------------------------------------------------
// 3) RMSNorm + RoPE on q,k, in place on bf16 [s,B,H,N,64] (s=0,1 only).
//  One wave per head-row: lane holds d=lane and d=lane+32 (rotate halves).
// ---------------------------------------------------------------------------
__global__ __launch_bounds__(256) void rmsrope_kernel(
    unsigned short* __restrict__ qkv,
    const float* __restrict__ qnw, const float* __restrict__ knw,
    const float* __restrict__ rcos, const float* __restrict__ rsin)
{
  const int lane = threadIdx.x & 31;
  const int row  = blockIdx.x * 8 + (threadIdx.x >> 5); // over 2*B*H*N = 2^18 rows
  const int s    = row >> 17;                           // 0 = q, 1 = k
  const int rem  = row & 131071;
  const int bh   = rem >> 11;
  const int t    = rem & 2047;
  unsigned short* p = qkv + (size_t)(s * 64 + bh) * 131072 + (size_t)t * 64;
  const float* wn = (s == 0) ? qnw : knw;

  float x1 = bf2f(p[lane]);
  float x2 = bf2f(p[lane + 32]);
  float ss = x1 * x1 + x2 * x2;
#pragma unroll
  for (int msk = 1; msk < 32; msk <<= 1) ss += __shfl_xor(ss, msk, 32);
  const float r   = rsqrtf(ss * (1.0f / 64.0f) + 1e-6f);
  const float xn1 = x1 * r * wn[lane];
  const float xn2 = x2 * r * wn[lane + 32];
  const float c1 = rcos[t * 64 + lane],      s1 = rsin[t * 64 + lane];
  const float c2 = rcos[t * 64 + lane + 32], s2 = rsin[t * 64 + lane + 32];
  p[lane]      = f2bf(xn1 * c1 - xn2 * s1);   // rot low  = -x2
  p[lane + 32] = f2bf(xn2 * c2 + xn1 * s2);   // rot high =  x1
}

// ---------------------------------------------------------------------------
// 4) Flash attention, bf16 WMMA, double-buffered async-to-LDS K/V^T tiles.
//  One block per (b,h, 128 q rows); 8 waves, each wave owns 16 q rows.
//  K tile: [32 key][64 d] row-major; V tile comes pre-transposed [64 d][32 key].
//  Output: bf16 [B, N, H*64] (proj-ready layout).
// ---------------------------------------------------------------------------
__global__ __launch_bounds__(256) void attn_kernel(
    const unsigned short* __restrict__ qkv,
    unsigned short* __restrict__ out)
{
  __shared__ unsigned short Kt [2][32 * 64];
  __shared__ unsigned short VtT[2][64 * 32];
  __shared__ unsigned short Pst[8 * 16 * 32];

  const int tid  = threadIdx.x;
  const int lane = tid & 31, wave = tid >> 5;
  const int half = lane >> 4, nl = lane & 15, kA = half * 8;
  const int qb = blockIdx.x & 15;               // 16 q-blocks of 128
  const int bh = blockIdx.x >> 4;               // b*16 + h

  const size_t hs = 131072;                     // N*D per (b,h)
  const unsigned short* Q   = qkv + (size_t)bh * hs;
  const unsigned short* Kp  = qkv + (size_t)(64 + bh) * hs;
  const unsigned short* VTp = qkv + (size_t)128 * hs + (size_t)bh * hs; // [d][t]

  // Q fragments (16 rows x 64 d), loaded once, A-layout.
  const int qrow = qb * 128 + wave * 16 + nl;
  const unsigned short* Qr = Q + (size_t)qrow * 64;
  FragU aq0, aq1;
  aq0.u[0] = *(const U4*)(Qr + kA);       aq0.u[1] = *(const U4*)(Qr + kA + 16);
  aq1.u[0] = *(const U4*)(Qr + 32 + kA);  aq1.u[1] = *(const U4*)(Qr + 32 + kA + 16);

  v8f zero = {};
  v8f o[4] = { zero, zero, zero, zero };
  float m[8], l[8];
#pragma unroll
  for (int i = 0; i < 8; ++i) { m[i] = -1e30f; l[i] = 0.0f; }
  const float scale = 0.125f;                   // 1/sqrt(64)

  const int krow = tid >> 3, kch = (tid & 7) * 8;   // K tile: 32 rows x 8 chunks
  const int vrow = tid >> 2, vch = (tid & 3) * 8;   // V^T tile: 64 rows x 4 chunks

  auto issue = [&](int j0, int buf) {
    async_b128(&Kt[buf][krow * 64 + kch],  Kp  + (size_t)(j0 + krow) * 64 + kch);
    async_b128(&VtT[buf][vrow * 32 + vch], VTp + (size_t)vrow * 2048 + j0 + vch);
  };

  issue(0, 0);
  int cur = 0;
  for (int j0 = 0; j0 < 2048; j0 += 32) {
    __syncthreads();                       // readers of buf^1 (prev iter) done
    if (j0 + 32 < 2048) {
      issue(j0 + 32, cur ^ 1);             // prefetch next key tile
      WAIT_ASYNC(0x2);                     // current tile (2 ops) complete
    } else {
      WAIT_ASYNC(0x0);
    }
    __syncthreads();                       // publish buf[cur]

    // S = Q Kt^T : two 16x16 tiles (key groups), K-dim 64 = 2 wmma each
    v8f s0 = zero, s1 = zero;
    {
      FragU b;
      b.u[0] = *(const U4*)&Kt[cur][nl * 64 + kA];
      b.u[1] = *(const U4*)&Kt[cur][nl * 64 + kA + 16];
      s0 = bf16_wmma(aq0, b, s0);
      b.u[0] = *(const U4*)&Kt[cur][nl * 64 + 32 + kA];
      b.u[1] = *(const U4*)&Kt[cur][nl * 64 + 32 + kA + 16];
      s0 = bf16_wmma(aq1, b, s0);
      b.u[0] = *(const U4*)&Kt[cur][(16 + nl) * 64 + kA];
      b.u[1] = *(const U4*)&Kt[cur][(16 + nl) * 64 + kA + 16];
      s1 = bf16_wmma(aq0, b, s1);
      b.u[0] = *(const U4*)&Kt[cur][(16 + nl) * 64 + 32 + kA];
      b.u[1] = *(const U4*)&Kt[cur][(16 + nl) * 64 + 32 + kA + 16];
      s1 = bf16_wmma(aq1, b, s1);
    }

    // online softmax over the 32-key tile (C-layout: reg i -> row i+8*half)
    float p0[8], p1[8];
#pragma unroll
    for (int i = 0; i < 8; ++i) {
      const float t0 = s0[i] * scale, t1 = s1[i] * scale;
      float mx = fmaxf(t0, t1);
#pragma unroll
      for (int msk = 1; msk < 16; msk <<= 1) mx = fmaxf(mx, __shfl_xor(mx, msk, 32));
      const float mn = fmaxf(m[i], mx);
      const float al = __expf(m[i] - mn);
      p0[i] = __expf(t0 - mn);
      p1[i] = __expf(t1 - mn);
      float rs = p0[i] + p1[i];
#pragma unroll
      for (int msk = 1; msk < 16; msk <<= 1) rs += __shfl_xor(rs, msk, 32);
      l[i] = l[i] * al + rs;
      m[i] = mn;
#pragma unroll
      for (int dt = 0; dt < 4; ++dt) o[dt][i] *= al;
    }

    // re-layout P (C-layout f32) -> A-layout bf16 via per-wave LDS staging
    const int pb = wave * 512;
#pragma unroll
    for (int i = 0; i < 8; ++i) {
      const int rl = i + 8 * half;
      Pst[pb + rl * 32 + nl]      = f2bf(p0[i]);
      Pst[pb + rl * 32 + 16 + nl] = f2bf(p1[i]);
    }
    FragU pa;
    pa.u[0] = *(const U4*)&Pst[pb + nl * 32 + kA];
    pa.u[1] = *(const U4*)&Pst[pb + nl * 32 + kA + 16];

    // O += P @ V : 4 d-tiles of 16
#pragma unroll
    for (int dt = 0; dt < 4; ++dt) {
      FragU vb;
      vb.u[0] = *(const U4*)&VtT[cur][(dt * 16 + nl) * 32 + kA];
      vb.u[1] = *(const U4*)&VtT[cur][(dt * 16 + nl) * 32 + kA + 16];
      o[dt] = bf16_wmma(pa, vb, o[dt]);
    }
    cur ^= 1;
  }

  // normalize and store bf16 [B, N, H*64]
  const int b = bh >> 4, h = bh & 15;
#pragma unroll
  for (int dt = 0; dt < 4; ++dt) {
#pragma unroll
    for (int i = 0; i < 8; ++i) {
      const int t = qb * 128 + wave * 16 + i + 8 * half;
      out[(size_t)(b * 2048 + t) * 1024 + h * 64 + dt * 16 + nl] = f2bf(o[dt][i] / l[i]);
    }
  }
}

// ---------------------------------------------------------------------------
// Launch
// ---------------------------------------------------------------------------
extern "C" void kernel_launch(void* const* d_in, const int* in_sizes, int n_in,
                              void* d_out, int out_size, void* d_ws, size_t ws_size,
                              hipStream_t stream) {
  const float* x        = (const float*)d_in[0];
  const float* rope_cos = (const float*)d_in[1];
  const float* rope_sin = (const float*)d_in[2];
  const float* qkv_w    = (const float*)d_in[3];
  const float* qkv_b    = (const float*)d_in[4];
  const float* qnorm_w  = (const float*)d_in[5];
  const float* knorm_w  = (const float*)d_in[6];
  const float* proj_w   = (const float*)d_in[7];
  const float* proj_b   = (const float*)d_in[8];
  float* out = (float*)d_out;

  // workspace layout (bytes); xb region is reused for attention output
  char* ws = (char*)d_ws;
  unsigned short* xb     = (unsigned short*)(ws);              // 16 MiB (x bf16, later attn out)
  unsigned short* wqkvb  = (unsigned short*)(ws + 16777216);   //  6 MiB
  unsigned short* wprojb = (unsigned short*)(ws + 23068672);   //  2 MiB
  unsigned short* qkvbuf = (unsigned short*)(ws + 25165824);   // 48 MiB q,k:[B,H,N,64], vT:[B,H,64,N]
  unsigned short* aout   = xb;                                 // reuse after GEMM1

  cvt_f32_bf16<<<8388608 / 256, 256, 0, stream>>>(x, xb, 8388608);
  cvt_f32_bf16<<<3145728 / 256, 256, 0, stream>>>(qkv_w, wqkvb, 3145728);
  cvt_f32_bf16<<<1048576 / 256, 256, 0, stream>>>(proj_w, wprojb, 1048576);

  dim3 g1(3072 / 128, 8192 / 64);
  gemm_bf16_kernel<1><<<g1, 256, 0, stream>>>(xb, wqkvb, qkv_b, qkvbuf, 1024, 3072);

  rmsrope_kernel<<<262144 / 8, 256, 0, stream>>>(qkvbuf, qnorm_w, knorm_w, rope_cos, rope_sin);

  attn_kernel<<<1024, 256, 0, stream>>>(qkvbuf, aout);

  dim3 g2(1024 / 128, 8192 / 64);
  gemm_bf16_kernel<0><<<g2, 256, 0, stream>>>(aout, wprojb, proj_b, out, 1024, 1024);
}
